// SchNetStudent_36661840839108
// MI455X (gfx1250) — matrix operand order repaired and checked by reading, not compiled, not run on hardware
//
#include <hip/hip_runtime.h>
#include <hip/hip_bf16.h>

#define NATOMS 20000
#define NPG    40
#define NGRAPH 500
#define DEG    16
#define NEDGE  320000
#define HID    128
#define NT     1250      // node row-tiles of 16 (20000/16)
#define STP    136       // per-wave staging pitch in halfs (16B-aligned, padded)

typedef __attribute__((ext_vector_type(16))) _Float16 v16h;
typedef __attribute__((ext_vector_type(8)))  float    v8f;
typedef __attribute__((ext_vector_type(4)))  _Float16 v4h;
union H16 { v16h v; uint4 q[2]; };

#define DEVFN static __device__ __forceinline__

DEVFN float sspf(float x) {
  // softplus(x) - ln2
  float sp = (x > 20.0f) ? x : log1pf(__expf(x));
  return sp - 0.69314718055994530942f;
}

DEVFN v8f wmma16(v16h a, v16h b, v8f c) {
  return __builtin_amdgcn_wmma_f32_16x16x32_f16(false, a, false, b, (short)0, c,
                                                false, false);
}

// A-fragment (16x32 f16) from f16 row-major [16][128] tile.
// lane<16: j=0..7 -> K=kb*32+0..7, j=8..15 -> K=kb*32+16..23 ; lane>=16: +8.
DEVFN v16h loadA_g16(const _Float16* base, int kb, int lane) {
  int m = lane & 15, hi = lane >> 4;
  const _Float16* p = base + m * HID + kb * 32 + hi * 8;
  H16 u;
  u.q[0] = *(const uint4*)(p);
  u.q[1] = *(const uint4*)(p + 16);
  return u.v;
}

// Same fragment pattern from per-wave LDS staging (pitch STP halfs).
DEVFN v16h loadA_lds16(const _Float16* stg, int kb, int lane) {
  int m = lane & 15, hi = lane >> 4;
  const _Float16* p = stg + m * STP + kb * 32 + hi * 8;
  H16 u;
  u.q[0] = *(const uint4*)(p);
  u.q[1] = *(const uint4*)(p + 16);
  return u.v;
}

// A-fragment from f32 row-major [16][128] tile (convert to f16 on load).
DEVFN v16h loadA_f32(const float* base, int kb, int lane) {
  int m = lane & 15, hi = lane >> 4;
  const float* p = base + m * HID + kb * 32 + hi * 8;
  float4 f0 = *(const float4*)(p);
  float4 f1 = *(const float4*)(p + 4);
  float4 f2 = *(const float4*)(p + 16);
  float4 f3 = *(const float4*)(p + 20);
  v16h a;
  a[0]=(_Float16)f0.x; a[1]=(_Float16)f0.y; a[2]=(_Float16)f0.z; a[3]=(_Float16)f0.w;
  a[4]=(_Float16)f1.x; a[5]=(_Float16)f1.y; a[6]=(_Float16)f1.z; a[7]=(_Float16)f1.w;
  a[8]=(_Float16)f2.x; a[9]=(_Float16)f2.y; a[10]=(_Float16)f2.z; a[11]=(_Float16)f2.w;
  a[12]=(_Float16)f3.x; a[13]=(_Float16)f3.y; a[14]=(_Float16)f3.z; a[15]=(_Float16)f3.w;
  return a;
}

// B-fragment (32x16 f16): weights pre-swizzled in LDS so each lane's 16
// elements are contiguous -> two ds_load_b128.
// storage index = (((kb*8+nb)*2+hi)*16 + nlo)*16 + j ; K=kb*32+8*hi+(j&7)+16*(j>>3), N=nb*16+nlo
DEVFN v16h loadB_lds(const _Float16* sW, int kb, int nb, int lane) {
  int nlo = lane & 15, hi = lane >> 4;
  const _Float16* p = sW + ((((kb * 8 + nb) * 2 + hi) * 16 + nlo) << 4);
  H16 u;
  u.q[0] = *(const uint4*)(p);
  u.q[1] = *(const uint4*)(p + 8);
  return u.v;
}

// Cooperative fill of a 128x128 f32 weight into LDS in B-fragment order (f16).
DEVFN void fillB(_Float16* sW, const float* __restrict__ w, int tid) {
  for (int i = tid; i < HID * HID; i += 256) {
    int j   = i & 15;
    int nlo = (i >> 4) & 15;
    int hi  = (i >> 8) & 1;
    int nb  = (i >> 9) & 7;
    int kb  = (i >> 12) & 3;
    int k = kb * 32 + 8 * hi + (j & 7) + 16 * (j >> 3);
    int n = nb * 16 + nlo;
    sW[i] = (_Float16)w[k * HID + n];
  }
}

// ---------------------------------------------------------------- h = emb[z]
__global__ void __launch_bounds__(256)
k_hinit(const float* __restrict__ emb, const int* __restrict__ z,
        float* __restrict__ h) {
  int i = blockIdx.x * 256 + threadIdx.x;
  if (i < NATOMS * HID) {
    int a = i >> 7, k = i & 127;
    h[i] = emb[z[a] * HID + k];
  }
}

// ------------------------------------- per-edge distance, cutoff, Gaussians
__global__ void __launch_bounds__(256)
k_edgeattr(const float* __restrict__ pos, const int* __restrict__ ei,
           _Float16* __restrict__ ea, float* __restrict__ Cv) {
  int e = blockIdx.x * 8 + (threadIdx.x >> 5);   // one edge per 32 lanes
  int lane = threadIdx.x & 31;
  int r = ei[e], c = ei[NEDGE + e];
  float dx = pos[r * 3 + 0] - pos[c * 3 + 0];
  float dy = pos[r * 3 + 1] - pos[c * 3 + 1];
  float dz = pos[r * 3 + 2] - pos[c * 3 + 2];
  float d = sqrtf(dx * dx + dy * dy + dz * dz);
  if (lane == 0) Cv[e] = 0.5f * (__cosf(d * 0.31415926535897931f) + 1.0f);
  const float delta = 10.0f / 127.0f;
  const float coeff = -0.5f / (delta * delta);
  v4h o;
#pragma unroll
  for (int t = 0; t < 4; t++) {
    int k = lane * 4 + t;
    float u = d - (float)k * delta;
    o[t] = (_Float16)__expf(coeff * u * u);
  }
  *(v4h*)(ea + (size_t)e * HID + lane * 4) = o;
}

// ------------------------------------------------- x1 = h @ cf_w1[l] (f32 out)
__global__ void __launch_bounds__(256)
k_x1(const float* __restrict__ h, const float* __restrict__ w,
     float* __restrict__ x1) {
  __shared__ _Float16 sW[HID * HID];
  int tid = threadIdx.x, lane = tid & 31, wave = tid >> 5;
  fillB(sW, w, tid);
  __syncthreads();
  int tile = blockIdx.x * 8 + wave;
  if (tile >= NT) return;
  int nlo = lane & 15, hi = lane >> 4;
  const float* hb = h + (size_t)tile * 16 * HID;
  v16h a0 = loadA_f32(hb, 0, lane), a1 = loadA_f32(hb, 1, lane);
  v16h a2 = loadA_f32(hb, 2, lane), a3 = loadA_f32(hb, 3, lane);
#pragma unroll
  for (int nb = 0; nb < 8; nb++) {
    v8f c;
#pragma unroll
    for (int r = 0; r < 8; r++) c[r] = 0.0f;   // cf_w1 has no bias
    c = wmma16(a0, loadB_lds(sW, 0, nb, lane), c);
    c = wmma16(a1, loadB_lds(sW, 1, nb, lane), c);
    c = wmma16(a2, loadB_lds(sW, 2, nb, lane), c);
    c = wmma16(a3, loadB_lds(sW, 3, nb, lane), c);
    float* o = x1 + (size_t)(tile * 16 + hi * 8) * HID + nb * 16 + nlo;
#pragma unroll
    for (int r = 0; r < 8; r++) o[r * HID] = c[r];
  }
}

// ---------- per-molecule edge pipeline: Wf = ssp(ea@W1+b1)@W2+b2; msg scatter
__global__ void __launch_bounds__(256)
k_edge(const _Float16* __restrict__ ea, const float* __restrict__ Cv,
       const float* __restrict__ x1, const int* __restrict__ col,
       const float* __restrict__ w1, const float* __restrict__ b1,
       const float* __restrict__ w2, const float* __restrict__ b2,
       float* __restrict__ aggG) {
  extern __shared__ char smem[];
  _Float16* sW1 = (_Float16*)smem;                   // 32 KB
  _Float16* sW2 = sW1 + HID * HID;                   // 32 KB
  float*    sB1 = (float*)(sW2 + HID * HID);         // 512 B
  float*    sB2 = sB1 + HID;                         // 512 B
  float*    sAgg = sB2 + HID;                        // 40*128*4 = 20 KB
  _Float16* sStg = (_Float16*)(sAgg + NPG * HID);    // 8 waves * 16*STP halfs

  int tid = threadIdx.x, lane = tid & 31, wave = tid >> 5;
  int g = blockIdx.x;
  fillB(sW1, w1, tid);
  fillB(sW2, w2, tid);
  for (int i = tid; i < HID; i += 256) { sB1[i] = b1[i]; sB2[i] = b2[i]; }
  for (int i = tid; i < NPG * HID; i += 256) sAgg[i] = 0.0f;
  __syncthreads();

  int nlo = lane & 15, hi = lane >> 4;
  _Float16* stg = sStg + wave * (16 * STP);

  for (int it = 0; it < 5; it++) {
    int t = wave * 5 + it;                 // tile within molecule (0..39)
    int ebase = g * 640 + t * 16;          // 16 edges, all sharing src row
    int rowatom = g * NPG + t;             // src atom for this tile

    const _Float16* A = ea + (size_t)ebase * HID;
    v16h a0 = loadA_g16(A, 0, lane), a1 = loadA_g16(A, 1, lane);
    v16h a2 = loadA_g16(A, 2, lane), a3 = loadA_g16(A, 3, lane);

    // GEMM1 + bias + ssp -> f16 staging (per-wave)
#pragma unroll
    for (int nb = 0; nb < 8; nb++) {
      float bv = sB1[nb * 16 + nlo];
      v8f c;
#pragma unroll
      for (int r = 0; r < 8; r++) c[r] = bv;
      c = wmma16(a0, loadB_lds(sW1, 0, nb, lane), c);
      c = wmma16(a1, loadB_lds(sW1, 1, nb, lane), c);
      c = wmma16(a2, loadB_lds(sW1, 2, nb, lane), c);
      c = wmma16(a3, loadB_lds(sW1, 3, nb, lane), c);
      _Float16* sp = stg + (hi * 8) * STP + nb * 16 + nlo;
#pragma unroll
      for (int r = 0; r < 8; r++) sp[r * STP] = (_Float16)sspf(c[r]);
    }
    asm volatile("s_wait_dscnt 0" ::: "memory");  // staging RAW (in-wave DS order)

    v16h e0 = loadA_lds16(stg, 0, lane), e1 = loadA_lds16(stg, 1, lane);
    v16h e2 = loadA_lds16(stg, 2, lane), e3 = loadA_lds16(stg, 3, lane);

    float cw[8]; int lc[8];
#pragma unroll
    for (int r = 0; r < 8; r++) {
      int e = ebase + hi * 8 + r;
      cw[r] = Cv[e];
      lc[r] = col[e] - g * NPG;            // local dst atom (0..39)
    }

    // GEMM2 + bias, * cutoff * x1[src], scatter-add into LDS agg
#pragma unroll
    for (int nb = 0; nb < 8; nb++) {
      float bv = sB2[nb * 16 + nlo];
      v8f c;
#pragma unroll
      for (int r = 0; r < 8; r++) c[r] = bv;
      c = wmma16(e0, loadB_lds(sW2, 0, nb, lane), c);
      c = wmma16(e1, loadB_lds(sW2, 1, nb, lane), c);
      c = wmma16(e2, loadB_lds(sW2, 2, nb, lane), c);
      c = wmma16(e3, loadB_lds(sW2, 3, nb, lane), c);
      float xv = x1[(size_t)rowatom * HID + nb * 16 + nlo];
#pragma unroll
      for (int r = 0; r < 8; r++) {
        float m = c[r] * cw[r] * xv;
        atomicAdd(&sAgg[lc[r] * HID + nb * 16 + nlo], m);
      }
    }
  }
  __syncthreads();
  float* og = aggG + (size_t)g * NPG * HID;
  for (int i = tid; i < NPG * HID; i += 256) og[i] = sAgg[i];
}

// ------------------ h += (ssp(agg@cf_w2+b2)) @ lin_w + lin_b  (two WMMA GEMMs)
__global__ void __launch_bounds__(256)
k_update(const float* __restrict__ agg, const float* __restrict__ w2,
         const float* __restrict__ b2, const float* __restrict__ lw,
         const float* __restrict__ lb, float* __restrict__ h) {
  extern __shared__ char smem[];
  _Float16* sW2 = (_Float16*)smem;
  _Float16* sLW = sW2 + HID * HID;
  float*    sB2 = (float*)(sLW + HID * HID);
  float*    sLB = sB2 + HID;
  _Float16* sStg = (_Float16*)(sLB + HID);

  int tid = threadIdx.x, lane = tid & 31, wave = tid >> 5;
  fillB(sW2, w2, tid);
  fillB(sLW, lw, tid);
  for (int i = tid; i < HID; i += 256) { sB2[i] = b2[i]; sLB[i] = lb[i]; }
  __syncthreads();

  int tile = blockIdx.x * 8 + wave;
  if (tile >= NT) return;
  int nlo = lane & 15, hi = lane >> 4;
  _Float16* stg = sStg + wave * (16 * STP);

  const float* ab = agg + (size_t)tile * 16 * HID;
  v16h a0 = loadA_f32(ab, 0, lane), a1 = loadA_f32(ab, 1, lane);
  v16h a2 = loadA_f32(ab, 2, lane), a3 = loadA_f32(ab, 3, lane);
#pragma unroll
  for (int nb = 0; nb < 8; nb++) {
    float bv = sB2[nb * 16 + nlo];
    v8f c;
#pragma unroll
    for (int r = 0; r < 8; r++) c[r] = bv;
    c = wmma16(a0, loadB_lds(sW2, 0, nb, lane), c);
    c = wmma16(a1, loadB_lds(sW2, 1, nb, lane), c);
    c = wmma16(a2, loadB_lds(sW2, 2, nb, lane), c);
    c = wmma16(a3, loadB_lds(sW2, 3, nb, lane), c);
    _Float16* sp = stg + (hi * 8) * STP + nb * 16 + nlo;
#pragma unroll
    for (int r = 0; r < 8; r++) sp[r * STP] = (_Float16)sspf(c[r]);
  }
  asm volatile("s_wait_dscnt 0" ::: "memory");

  v16h e0 = loadA_lds16(stg, 0, lane), e1 = loadA_lds16(stg, 1, lane);
  v16h e2 = loadA_lds16(stg, 2, lane), e3 = loadA_lds16(stg, 3, lane);
#pragma unroll
  for (int nb = 0; nb < 8; nb++) {
    float bv = sLB[nb * 16 + nlo];
    v8f c;
#pragma unroll
    for (int r = 0; r < 8; r++) c[r] = bv;
    c = wmma16(e0, loadB_lds(sLW, 0, nb, lane), c);
    c = wmma16(e1, loadB_lds(sLW, 1, nb, lane), c);
    c = wmma16(e2, loadB_lds(sLW, 2, nb, lane), c);
    c = wmma16(e3, loadB_lds(sLW, 3, nb, lane), c);
    float* o = h + (size_t)(tile * 16 + hi * 8) * HID + nb * 16 + nlo;
#pragma unroll
    for (int r = 0; r < 8; r++) o[r * HID] += c[r];   // residual
  }
}

// ----------------------- head MLP + per-graph sum (tiny: scalar, per molecule)
__global__ void __launch_bounds__(256)
k_head(const float* __restrict__ h, const float* __restrict__ w1,
       const float* __restrict__ b1, const float* __restrict__ w2,
       const float* __restrict__ b2, float* __restrict__ out) {
  __shared__ float sW1[HID * 64];
  __shared__ float sH[NPG * 132];
  __shared__ float sB1[64];
  __shared__ float sW2[64];
  __shared__ float sY[NPG];
  int g = blockIdx.x, tid = threadIdx.x;
  for (int i = tid; i < HID * 64; i += 256) sW1[i] = w1[i];
  for (int i = tid; i < 64; i += 256) { sB1[i] = b1[i]; sW2[i] = w2[i]; }
  for (int i = tid; i < NPG * HID; i += 256) {
    int a = i >> 7, k = i & 127;
    sH[a * 132 + k] = h[(size_t)(g * NPG + a) * HID + k];
  }
  __syncthreads();
  if (tid < NPG) {
    const float* hr = sH + tid * 132;
    float y = b2[0];
    for (int j = 0; j < 64; j++) {
      float acc = sB1[j];
#pragma unroll 4
      for (int k = 0; k < HID; k++) acc += hr[k] * sW1[k * 64 + j];
      y += sspf(acc) * sW2[j];
    }
    sY[tid] = y;
  }
  __syncthreads();
  if (tid == 0) {
    float s = 0.0f;
    for (int a = 0; a < NPG; a++) s += sY[a];
    out[g] = s;
  }
}

extern "C" void kernel_launch(void* const* d_in, const int* in_sizes, int n_in,
                              void* d_out, int out_size, void* d_ws, size_t ws_size,
                              hipStream_t stream) {
  const float* pos     = (const float*)d_in[0];
  const float* emb     = (const float*)d_in[1];
  const float* mlp_w1  = (const float*)d_in[2];
  const float* mlp_b1  = (const float*)d_in[3];
  const float* mlp_w2  = (const float*)d_in[4];
  const float* mlp_b2  = (const float*)d_in[5];
  const float* cf_w1   = (const float*)d_in[6];
  const float* cf_w2   = (const float*)d_in[7];
  const float* cf_b2   = (const float*)d_in[8];
  const float* lin_w   = (const float*)d_in[9];
  const float* lin_b   = (const float*)d_in[10];
  const float* head_w1 = (const float*)d_in[11];
  const float* head_b1 = (const float*)d_in[12];
  const float* head_w2 = (const float*)d_in[13];
  const float* head_b2 = (const float*)d_in[14];
  const int*   z       = (const int*)d_in[15];
  const int*   ei      = (const int*)d_in[17];
  float* out = (float*)d_out;

  char* ws = (char*)d_ws;
  _Float16* ea = (_Float16*)ws;  ws += (size_t)NEDGE * HID * sizeof(_Float16);
  float* Cv = (float*)ws;        ws += (size_t)NEDGE * sizeof(float);
  float* h  = (float*)ws;        ws += (size_t)NATOMS * HID * sizeof(float);
  float* x1 = (float*)ws;        ws += (size_t)NATOMS * HID * sizeof(float);
  float* agg = (float*)ws;       ws += (size_t)NATOMS * HID * sizeof(float);

  const size_t shEdge = (size_t)2 * HID * HID * 2 + 2 * HID * 4 +
                        (size_t)NPG * HID * 4 + (size_t)8 * 16 * STP * 2;
  const size_t shUpd  = (size_t)2 * HID * HID * 2 + 2 * HID * 4 +
                        (size_t)8 * 16 * STP * 2;
  (void)hipFuncSetAttribute((const void*)k_edge,
      hipFuncAttributeMaxDynamicSharedMemorySize, (int)shEdge);
  (void)hipFuncSetAttribute((const void*)k_update,
      hipFuncAttributeMaxDynamicSharedMemorySize, (int)shUpd);

  k_hinit<<<(NATOMS * HID + 255) / 256, 256, 0, stream>>>(emb, z, h);
  k_edgeattr<<<NEDGE / 8, 256, 0, stream>>>(pos, ei, ea, Cv);

  for (int l = 0; l < 3; l++) {
    k_x1<<<(NT + 7) / 8, 256, 0, stream>>>(h, cf_w1 + (size_t)l * HID * HID, x1);
    k_edge<<<NGRAPH, 256, shEdge, stream>>>(
        ea, Cv, x1, ei + NEDGE,
        mlp_w1 + (size_t)l * HID * HID, mlp_b1 + (size_t)l * HID,
        mlp_w2 + (size_t)l * HID * HID, mlp_b2 + (size_t)l * HID, agg);
    k_update<<<(NT + 7) / 8, 256, shUpd, stream>>>(
        agg, cf_w2 + (size_t)l * HID * HID, cf_b2 + (size_t)l * HID,
        lin_w + (size_t)l * HID * HID, lin_b + (size_t)l * HID, h);
  }
  k_head<<<NGRAPH, 256, 0, stream>>>(h, head_w1, head_b1, head_w2, head_b2, out);
}
